// RealNVP_80573586473693
// MI455X (gfx1250) — matrix operand order repaired and checked by reading, not compiled, not run on hardware
//
#include <hip/hip_runtime.h>
#include <hip/hip_bf16.h>

// ---------------------------------------------------------------------------
// RealNVP forward flow for MI455X (gfx1250, wave32, WMMA 16x16x32 bf16)
//
// Compute-bound (155 GFLOP vs ~25 MB HBM traffic). All three matmuls of each
// coupling MLP run on v_wmma_f32_16x16x32_bf16. Weights are converted once
// per launch to bf16 in d_ws, pre-swizzled into the CDNA5 B-fragment lane
// layout (2x global_load_b128 per fragment, fully coalesced). Activations
// stay in LDS row-major bf16; CDNA5's A layout reads contiguous K-octets so
// an A fragment is 2x ds_load_b128 per lane.
//
// MTILE=64 rows x 16 waves per workgroup: each B fragment feeds 4 M-tile
// WMMAs -> 64 FLOP per L2 byte of weight traffic, keeping the WMMA pipes fed
// from the 192 MB L2 (all 9.4 MB of packed weights resident, reused by all
// 256 workgroups).
// ---------------------------------------------------------------------------

typedef __bf16 bf16;
typedef __attribute__((ext_vector_type(8)))  __bf16 v8bf;
typedef __attribute__((ext_vector_type(16))) __bf16 v16bf;
typedef __attribute__((ext_vector_type(8)))  float  v8f;

#define N_DIMS   64
#define N_HALF   32
#define HIDDEN   512
#define N_LAYERS 8
#define BATCH    16384
#define MTILE    64          // batch rows per workgroup
#define NTHREADS 512
#define NWAVES   16

// packed weight block (bf16 elements) per (layer, mlp)
#define W1_ELEMS (32 * 512)          // 16384
#define W2_ELEMS (512 * 512)         // 262144
#define W3_ELEMS (512 * 32)          // 16384
#define BLK_ELEMS (W1_ELEMS + W2_ELEMS + W3_ELEMS)   // 294912
#define W2_OFF   W1_ELEMS
#define W3_OFF   (W1_ELEMS + W2_ELEMS)

// ---------------------------------------------------------------------------
// Weight pack: fp32 [L][K][N] row-major  ->  bf16 WMMA B fragments.
// Fragment (nt, kt) covers K = kt*32..+31, N = nt*16..+15.
// Lane L (0..31): n = nt*16 + (L&15), khalf = L>>4;
//   16 values j=0..15 are W[kt*32 + (j<8 ? khalf*8+j : 16+khalf*8+(j-8))][n],
//   stored contiguously (same K-octet order as the A fragment).
// ---------------------------------------------------------------------------
__global__ __launch_bounds__(256) void pack_weights_kernel(
    const float* __restrict__ tW1, const float* __restrict__ tW2,
    const float* __restrict__ tW3, const float* __restrict__ sW1,
    const float* __restrict__ sW2, const float* __restrict__ sW3,
    bf16* __restrict__ wpack)
{
    int f     = blockIdx.x;       // 0 .. 16*576-1
    int blk   = f / 576;          // layer*2 + mlp
    int fi    = f % 576;
    int layer = blk >> 1;
    int mlp   = blk & 1;

    const float* W;
    int nt, kt, kdim, ndim;
    long roff;
    if (fi < 32) {                // W1: 32x512, single K-tile
        nt = fi; kt = 0; W = mlp ? sW1 : tW1; kdim = 32;  ndim = 512;
        roff = (long)nt * 512;
    } else if (fi < 544) {        // W2: 512x512
        int q = fi - 32;  nt = q >> 4; kt = q & 15;
        W = mlp ? sW2 : tW2; kdim = 512; ndim = 512;
        roff = (long)W2_OFF + ((long)nt * 16 + kt) * 512;
    } else {                      // W3: 512x32
        int q = fi - 544; nt = q >> 4; kt = q & 15;
        W = mlp ? sW3 : tW3; kdim = 512; ndim = 32;
        roff = (long)W3_OFF + ((long)nt * 16 + kt) * 512;
    }
    const float* Wl = W + (long)layer * kdim * ndim;
    bf16* dst = wpack + (long)blk * BLK_ELEMS + roff;

    for (int e = threadIdx.x; e < 512; e += 256) {
        int lane = e >> 4, j = e & 15;
        int nl = lane & 15, khalf = lane >> 4;
        int k = kt * 32 + (j < 8 ? khalf * 8 + j : 16 + khalf * 8 + (j - 8));
        int n = nt * 16 + nl;
        dst[e] = (bf16)Wl[(long)k * ndim + n];
    }
}

// A fragment: lane L holds row M=(L&15); half=(L>>4) selects K-octets:
// V0-3 = K base+half*8 .. +7, V4-7 = K base+16+half*8 .. +7 (contiguous 16B each).
__device__ inline v16bf load_a16(const bf16* p) {
    v8bf lo = *(const v8bf*)p;
    v8bf hi = *(const v8bf*)(p + 16);
    return __builtin_shufflevector(lo, hi,
        0, 1, 2, 3, 4, 5, 6, 7, 8, 9, 10, 11, 12, 13, 14, 15);
}

// ---------------------------------------------------------------------------
// One GEMM stage: [MTILE x (KTS*32)] @ [(KTS*32) x (NTCNT*16)] + bias,
// optional ReLU, output either bf16 (pitch HIDDEN) or fp32 (pitch N_HALF).
// 16 waves split the N tiles; each wave keeps 4 M-tile accumulators so every
// B fragment load is amortized over 4 WMMAs.
// ---------------------------------------------------------------------------
template<int KTS, int NTCNT, int APITCH, bool RELU, bool OUTF32>
__device__ inline void gemm_stage(int wave, int lane,
                                  const bf16* __restrict__ wbase,
                                  const float* __restrict__ bias,
                                  const bf16* A, bf16* outB, float* outF)
{
    const int nl = lane & 15, half = lane >> 4;
    for (int nt = wave; nt < NTCNT; nt += NWAVES) {
        float bv = bias[nt * 16 + nl];
        v8f acc[4];
#pragma unroll
        for (int mt = 0; mt < 4; ++mt)
#pragma unroll
            for (int e = 0; e < 8; ++e) acc[mt][e] = bv;

        const bf16* fragp = wbase + (long)nt * KTS * 512 + lane * 16;
        const bf16* arow  = A + nl * APITCH + half * 8;
        for (int kt = 0; kt < KTS; ++kt) {
            v16bf B = *(const v16bf*)fragp;         // 2x global_load_b128
            fragp += 512;
#pragma unroll
            for (int mt = 0; mt < 4; ++mt) {
                v16bf Af = load_a16(arow + mt * 16 * APITCH + kt * 32);
                acc[mt] = __builtin_amdgcn_wmma_f32_16x16x32_bf16(
                              false, Af, false, B, (short)0, acc[mt], false, false);
            }
        }

        int col = nt * 16 + nl;
#pragma unroll
        for (int mt = 0; mt < 4; ++mt) {
            int rbase = mt * 16 + half * 8;          // C layout: M = r + 8*(lane>>4)
#pragma unroll
            for (int r = 0; r < 8; ++r) {
                float v = acc[mt][r];
                if (RELU) v = fmaxf(v, 0.f);
                if (OUTF32) outF[(rbase + r) * N_HALF + col] = v;
                else        outB[(rbase + r) * HIDDEN + col] = (bf16)v;
            }
        }
    }
}

__global__ __launch_bounds__(NTHREADS) void realnvp_flow_kernel(
    const float* __restrict__ y,
    const float* __restrict__ tb1, const float* __restrict__ tb2,
    const float* __restrict__ tb3, const float* __restrict__ sb1,
    const float* __restrict__ sb2, const float* __restrict__ sb3,
    const float* __restrict__ s_scale, const float* __restrict__ s_shift,
    const bf16* __restrict__ wpack,
    float* __restrict__ z_out, float* __restrict__ ld_out)
{
    __shared__ float z_s [MTILE][N_DIMS];      // 16 KB : flow state
    __shared__ float ld_s[MTILE][N_DIMS];      // 16 KB : log-det accumulator
    __shared__ bf16  z1bf[MTILE][N_HALF];      //  4 KB : MLP input (bf16)
    __shared__ bf16  h1  [MTILE][HIDDEN];      // 64 KB : activations
    __shared__ bf16  h2  [MTILE][HIDDEN];      // 64 KB
    __shared__ float ts  [2][MTILE][N_HALF];   // 16 KB : t and raw-s tiles

    const int tid  = threadIdx.x;
    const int wave = tid >> 5, lane = tid & 31;
    const int row0 = blockIdx.x * MTILE;

    for (int e = tid; e < MTILE * N_DIMS; e += NTHREADS) {
        int r = e >> 6, d = e & 63;
        z_s[r][d]  = y[(long)(row0 + r) * N_DIMS + d];
        ld_s[r][d] = 0.f;
    }
    __syncthreads();

    for (int i = 0; i < N_LAYERS; ++i) {
        const int odd = i & 1;
        // z1 = z[:, i1] in bf16 (i even: odd dims; i odd: even dims)
        for (int e = tid; e < MTILE * N_HALF; e += NTHREADS) {
            int r = e >> 5, j = e & 31;
            z1bf[r][j] = (bf16)z_s[r][2 * j + (1 - odd)];
        }
        __syncthreads();

        for (int mlp = 0; mlp < 2; ++mlp) {     // 0 = t-net, 1 = s-net
            const bf16* blk = wpack + (long)(i * 2 + mlp) * BLK_ELEMS;
            const float* b1 = (mlp ? sb1 : tb1) + i * HIDDEN;
            const float* b2 = (mlp ? sb2 : tb2) + i * HIDDEN;
            const float* b3 = (mlp ? sb3 : tb3) + i * N_HALF;

            gemm_stage<1, 32, N_HALF, true, false>(wave, lane, blk, b1,
                &z1bf[0][0], &h1[0][0], nullptr);
            __syncthreads();
            gemm_stage<16, 32, HIDDEN, true, false>(wave, lane, blk + W2_OFF, b2,
                &h1[0][0], &h2[0][0], nullptr);
            __syncthreads();
            gemm_stage<16, 2, HIDDEN, false, true>(wave, lane, blk + W3_OFF, b3,
                &h2[0][0], nullptr, &ts[mlp][0][0]);
            __syncthreads();
        }

        // affine coupling: z0 = z0*exp(s)+t ; log_det[:, i0] += s
        for (int e = tid; e < MTILE * N_HALF; e += NTHREADS) {
            int r = e >> 5, j = e & 31;
            float sf = tanhf(ts[1][r][j]) * s_scale[i * N_HALF + j]
                     + s_shift[i * N_HALF + j];
            int d0 = 2 * j + odd;
            z_s[r][d0] = z_s[r][d0] * expf(sf) + ts[0][r][j];
            ld_s[r][d0] += sf;
        }
        __syncthreads();
    }

    for (int e = tid; e < MTILE * N_DIMS; e += NTHREADS) {
        int r = e >> 6, d = e & 63;
        z_out [(long)(row0 + r) * N_DIMS + d] = z_s[r][d];
        ld_out[(long)(row0 + r) * N_DIMS + d] = ld_s[r][d];
    }
}

extern "C" void kernel_launch(void* const* d_in, const int* in_sizes, int n_in,
                              void* d_out, int out_size, void* d_ws, size_t ws_size,
                              hipStream_t stream)
{
    const float* y       = (const float*)d_in[0];
    const float* tW1     = (const float*)d_in[1];
    const float* tb1     = (const float*)d_in[2];
    const float* tW2     = (const float*)d_in[3];
    const float* tb2     = (const float*)d_in[4];
    const float* tW3     = (const float*)d_in[5];
    const float* tb3     = (const float*)d_in[6];
    const float* sW1     = (const float*)d_in[7];
    const float* sb1     = (const float*)d_in[8];
    const float* sW2     = (const float*)d_in[9];
    const float* sb2     = (const float*)d_in[10];
    const float* sW3     = (const float*)d_in[11];
    const float* sb3     = (const float*)d_in[12];
    const float* s_scale = (const float*)d_in[13];
    const float* s_shift = (const float*)d_in[14];

    bf16*  wpack  = (bf16*)d_ws;                 // 9.44 MB packed bf16 weights
    float* z_out  = (float*)d_out;
    float* ld_out = z_out + (long)BATCH * N_DIMS;

    pack_weights_kernel<<<16 * 576, 256, 0, stream>>>(
        tW1, tW2, tW3, sW1, sW2, sW3, wpack);

    realnvp_flow_kernel<<<BATCH / MTILE, NTHREADS, 0, stream>>>(
        y, tb1, tb2, tb3, sb1, sb2, sb3, s_scale, s_shift,
        wpack, z_out, ld_out);
}
